// GRU_82429012345450
// MI455X (gfx1250) — compile-verified, matrix operand order checked by reading, not run on hardware
//
#include <hip/hip_runtime.h>
#include <hip/hip_bf16.h>

// ---------------------------------------------------------------------------
// GRU on MI455X (gfx1250, wave32).
//   Phase 1: convert weights/X to bf16 (one pass, L2-resident afterwards)
//   Phase 2: WMMA bf16 GEMMs precompute Gx_g[t] = W_gx @ x_t + b_g  (g=u,r,c)
//   Phase 3: persistent kernel runs the serial recurrence (bandwidth-bound
//            bf16 matvecs from L2, grid barriers via atomic counter)
//   Phase 4: WMMA bf16 GEMM computes Y = W_y @ H + b_y for all t
// ---------------------------------------------------------------------------

#define SEQ 2048
#define DIN 1024
#define HID 2048
#define DOUT 1024

#define REC_BLOCKS 64
#define TPB 256          // GEMM / utility block size
#define TPB_REC 1024     // recurrent persistent block size (32 waves/block)

typedef __attribute__((ext_vector_type(16))) __bf16 v16bf;
typedef __attribute__((ext_vector_type(8)))  __bf16 v8bf;
typedef __attribute__((ext_vector_type(8)))  float  v8f;

// --------------------------------------------------------------------------
// fp32 -> bf16 slice conversion:  dst[r*cols + c] = src[r*srcld + colOff + c]
// --------------------------------------------------------------------------
__global__ void gru_convert_slice(const float* __restrict__ src, int srcld, int colOff,
                                  __bf16* __restrict__ dst, int rows, int cols) {
    size_t total = (size_t)rows * cols;
    for (size_t idx = (size_t)blockIdx.x * blockDim.x + threadIdx.x;
         idx < total; idx += (size_t)gridDim.x * blockDim.x) {
        size_t r = idx / cols;
        size_t c = idx - r * cols;
        dst[idx] = (__bf16)src[r * srcld + colOff + c];
    }
}

// --------------------------------------------------------------------------
// Zero h0 and the grid-barrier counter (runs every call -> deterministic)
// --------------------------------------------------------------------------
__global__ void gru_init(float* __restrict__ hbuf, unsigned* __restrict__ counter) {
    int i = blockIdx.x * blockDim.x + threadIdx.x;
    if (i < HID) hbuf[i] = 0.0f;
    if (i == 0) *counter = 0u;
}

// --------------------------------------------------------------------------
// bf16 WMMA GEMM:  C[n][m] = bias[m] + sum_k A[m][k] * B[n][k]
//   A: M x K row-major (bf16), B: N x K row-major (bf16),
//   C: N x M row-major (fp32). M % 128 == 0, N % 16 == 0, K % 32 == 0.
// One wave computes one 16x16 tile of C. 8 waves (256 thr) per block.
// Fragment layouts per CDNA5 ISA 7.12.2 (16-bit A 16x32; 32-bit C/D 16x16).
// --------------------------------------------------------------------------
__global__ __launch_bounds__(TPB) void gru_gemm_bf16(
    const __bf16* __restrict__ A, const __bf16* __restrict__ B,
    const float* __restrict__ bias, float* __restrict__ C,
    int M, int N, int K) {
    const int lane = threadIdx.x & 31;
    const int wave = threadIdx.x >> 5;
    const int m0 = (blockIdx.x * 8 + wave) * 16;
    const int n0 = blockIdx.y * 16;
    if (m0 >= M || n0 >= N) return;

    const int half = lane >> 4;   // 0 or 1
    const int l    = lane & 15;   // 0..15

    v8f acc = {};
    const __bf16* arow = A + (size_t)(m0 + l) * K + half * 8;
    const __bf16* brow = B + (size_t)(n0 + l) * K + half * 8;

    for (int k0 = 0; k0 < K; k0 += 32) {
        // A fragment: lane l holds row m0+l; halves split K into 8+8 chunks.
        v8bf a_lo = *(const v8bf*)(arow + k0);
        v8bf a_hi = *(const v8bf*)(arow + k0 + 16);
        v8bf b_lo = *(const v8bf*)(brow + k0);
        v8bf b_hi = *(const v8bf*)(brow + k0 + 16);
        v16bf a, b;
#pragma unroll
        for (int p = 0; p < 8; ++p) {
            a[p]     = a_lo[p];
            a[8 + p] = a_hi[p];
            b[p]     = b_lo[p];
            b[8 + p] = b_hi[p];
        }
        acc = __builtin_amdgcn_wmma_f32_16x16x32_bf16(
            /*neg_a=*/false, a, /*neg_b=*/false, b,
            /*c_mod=*/(short)0, acc, /*reuse_a=*/false, /*reuse_b=*/false);
    }

    // C/D layout: lane -> n = n0 + l ; element v -> m = m0 + 8*half + v
    const int mb = m0 + 8 * half;
    float* crow = C + (size_t)(n0 + l) * M + mb;
    float4 lo, hi;
    lo.x = acc[0] + bias[mb + 0];
    lo.y = acc[1] + bias[mb + 1];
    lo.z = acc[2] + bias[mb + 2];
    lo.w = acc[3] + bias[mb + 3];
    hi.x = acc[4] + bias[mb + 4];
    hi.y = acc[5] + bias[mb + 5];
    hi.z = acc[6] + bias[mb + 6];
    hi.w = acc[7] + bias[mb + 7];
    *(float4*)(crow)     = lo;
    *(float4*)(crow + 4) = hi;
}

// --------------------------------------------------------------------------
// Grid-wide barrier (monotonic counter, agent-scope fences, s_sleep spin)
// --------------------------------------------------------------------------
__device__ __forceinline__ void grid_barrier(unsigned* cnt, unsigned target) {
    __builtin_amdgcn_fence(__ATOMIC_RELEASE, "agent");
    __syncthreads();
    if (threadIdx.x == 0) {
        __hip_atomic_fetch_add(cnt, 1u, __ATOMIC_RELAXED, __HIP_MEMORY_SCOPE_AGENT);
        while (__hip_atomic_load(cnt, __ATOMIC_RELAXED, __HIP_MEMORY_SCOPE_AGENT) < target) {
            __builtin_amdgcn_s_sleep(2);
        }
    }
    __syncthreads();
    __builtin_amdgcn_fence(__ATOMIC_ACQUIRE, "agent");
}

// bf16 row (HID elems, packed 8/uint4) dotted against an LDS fp32 vector.
// Each lane streams 16B global_load_b128's (fully coalesced: 512B per wave
// iteration); bf16 -> f32 unpack is just a shift.  Wave-level shuffle reduce.
__device__ __forceinline__ float row_dot(const __bf16* __restrict__ row,
                                         const float* __restrict__ sv, int lane) {
    const uint4* w4 = (const uint4*)row;     // HID/8 = 256 uint4 per row
    float acc = 0.0f;
#pragma unroll
    for (int it = 0; it < HID / 8 / 32; ++it) {   // 8 iterations
        const int kk = it * 32 + lane;
        uint4 q = w4[kk];
        const float* s = sv + kk * 8;
        acc += __uint_as_float(q.x << 16)         * s[0];
        acc += __uint_as_float(q.x & 0xffff0000u) * s[1];
        acc += __uint_as_float(q.y << 16)         * s[2];
        acc += __uint_as_float(q.y & 0xffff0000u) * s[3];
        acc += __uint_as_float(q.z << 16)         * s[4];
        acc += __uint_as_float(q.z & 0xffff0000u) * s[5];
        acc += __uint_as_float(q.w << 16)         * s[6];
        acc += __uint_as_float(q.w & 0xffff0000u) * s[7];
    }
#pragma unroll
    for (int off = 16; off; off >>= 1) acc += __shfl_xor(acc, off, 32);
    return acc;
}

// --------------------------------------------------------------------------
// Persistent recurrent kernel: 64 blocks x 1024 threads (2048 waves).
// Per step:
//   A) u = sigmoid(Gu[t] + Wuh@h), r = sigmoid(Gr[t] + Wrh@h), rh = r*h
//   -- grid barrier --
//   B) c = tanh(Gc[t] + Wch@rh), h = u*c + (1-u)*h  -> hidden_out, Hbf
//   -- grid barrier --
// All weight reads are bf16 and L2-resident (24 MB/step, 49 GB total).
// --------------------------------------------------------------------------
__global__ __launch_bounds__(TPB_REC, 1) void gru_recurrent(
    const __bf16* __restrict__ Wuh, const __bf16* __restrict__ Wrh,
    const __bf16* __restrict__ Wch,
    const float* __restrict__ Gu, const float* __restrict__ Gr,
    const float* __restrict__ Gc,
    float* __restrict__ hbuf, float* __restrict__ ubuf, float* __restrict__ rhbuf,
    float* __restrict__ hidden_out, __bf16* __restrict__ Hbf,
    unsigned* __restrict__ counter) {
    __shared__ float sv[HID];    // staged h (part A) then rh (part B)

    const int tid    = threadIdx.x;
    const int lane   = tid & 31;
    const int wave   = tid >> 5;
    const int gwave  = blockIdx.x * (TPB_REC / 32) + wave;
    const int nwaves = REC_BLOCKS * (TPB_REC / 32);   // 2048 waves

    unsigned target = 0;

    for (int t = 0; t < SEQ; ++t) {
        // stage h into LDS
        for (int i = tid; i < HID; i += TPB_REC) sv[i] = hbuf[i];
        __syncthreads();

        // ---- part A: rows 0..2047 -> u, rows 2048..4095 -> r ----
        for (int row = gwave; row < 2 * HID; row += nwaves) {
            const bool isU = row < HID;
            const int  i   = isU ? row : row - HID;
            const __bf16* wrow = (isU ? Wuh : Wrh) + (size_t)i * HID;
            float acc = row_dot(wrow, sv, lane);
            if (lane == 0) {
                float g = isU ? Gu[(size_t)t * HID + i] : Gr[(size_t)t * HID + i];
                float s = 1.0f / (1.0f + __expf(-(acc + g)));
                if (isU) ubuf[i] = s;
                else     rhbuf[i] = s * sv[i];
            }
        }

        target += REC_BLOCKS;
        grid_barrier(counter, target);

        // stage rh into LDS
        for (int i = tid; i < HID; i += TPB_REC) sv[i] = rhbuf[i];
        __syncthreads();

        // ---- part B: candidate + state update ----
        for (int row = gwave; row < HID; row += nwaves) {
            const __bf16* wrow = Wch + (size_t)row * HID;
            float acc = row_dot(wrow, sv, lane);
            if (lane == 0) {
                float c    = tanhf(acc + Gc[(size_t)t * HID + row]);
                float u    = ubuf[row];
                float hold = hbuf[row];
                float hn   = u * c + (1.0f - u) * hold;
                hbuf[row] = hn;
                hidden_out[(size_t)t * HID + row] = hn;
                Hbf[(size_t)t * HID + row] = (__bf16)hn;
            }
        }

        target += REC_BLOCKS;
        grid_barrier(counter, target);
    }
}

// --------------------------------------------------------------------------
extern "C" void kernel_launch(void* const* d_in, const int* in_sizes, int n_in,
                              void* d_out, int out_size, void* d_ws, size_t ws_size,
                              hipStream_t stream) {
    const float* X   = (const float*)d_in[0];   // (SEQ, DIN)
    const float* W_u = (const float*)d_in[1];   // (HID, HID+DIN)
    const float* b_u = (const float*)d_in[2];
    const float* W_r = (const float*)d_in[3];
    const float* b_r = (const float*)d_in[4];
    const float* W_c = (const float*)d_in[5];
    const float* b_c = (const float*)d_in[6];
    const float* W_y = (const float*)d_in[7];   // (DOUT, HID)
    const float* b_y = (const float*)d_in[8];

    float* Y    = (float*)d_out;                     // (SEQ, DOUT)
    float* Hout = Y + (size_t)SEQ * DOUT;            // (SEQ, HID)

    // ---- workspace layout ----
    char* ws = (char*)d_ws;
    size_t off = 0;
    auto alloc = [&](size_t bytes) { void* p = ws + off; off += (bytes + 255) & ~(size_t)255; return p; };

    __bf16* Wuh = (__bf16*)alloc((size_t)HID * HID * 2);
    __bf16* Wrh = (__bf16*)alloc((size_t)HID * HID * 2);
    __bf16* Wch = (__bf16*)alloc((size_t)HID * HID * 2);
    __bf16* Wux = (__bf16*)alloc((size_t)HID * DIN * 2);
    __bf16* Wrx = (__bf16*)alloc((size_t)HID * DIN * 2);
    __bf16* Wcx = (__bf16*)alloc((size_t)HID * DIN * 2);
    __bf16* Wyb = (__bf16*)alloc((size_t)DOUT * HID * 2);
    __bf16* Xb  = (__bf16*)alloc((size_t)SEQ * DIN * 2);
    float*  Gu  = (float*) alloc((size_t)SEQ * HID * 4);
    float*  Gr  = (float*) alloc((size_t)SEQ * HID * 4);
    float*  Gc  = (float*) alloc((size_t)SEQ * HID * 4);
    __bf16* Hbf = (__bf16*)alloc((size_t)SEQ * HID * 2);
    float*  hbuf  = (float*)alloc(HID * 4);
    float*  ubuf  = (float*)alloc(HID * 4);
    float*  rhbuf = (float*)alloc(HID * 4);
    unsigned* counter = (unsigned*)alloc(256);
    (void)ws_size; (void)in_sizes; (void)n_in; (void)out_size;

    // ---- phase 1: bf16 conversions ----
    dim3 cb(256), cg(512);
    gru_convert_slice<<<cg, cb, 0, stream>>>(W_u, HID + DIN, 0,   Wuh, HID, HID);
    gru_convert_slice<<<cg, cb, 0, stream>>>(W_r, HID + DIN, 0,   Wrh, HID, HID);
    gru_convert_slice<<<cg, cb, 0, stream>>>(W_c, HID + DIN, 0,   Wch, HID, HID);
    gru_convert_slice<<<cg, cb, 0, stream>>>(W_u, HID + DIN, HID, Wux, HID, DIN);
    gru_convert_slice<<<cg, cb, 0, stream>>>(W_r, HID + DIN, HID, Wrx, HID, DIN);
    gru_convert_slice<<<cg, cb, 0, stream>>>(W_c, HID + DIN, HID, Wcx, HID, DIN);
    gru_convert_slice<<<cg, cb, 0, stream>>>(W_y, HID,       0,   Wyb, DOUT, HID);
    gru_convert_slice<<<cg, cb, 0, stream>>>(X,   DIN,       0,   Xb,  SEQ,  DIN);

    gru_init<<<(HID + 255) / 256, 256, 0, stream>>>(hbuf, counter);

    // ---- phase 2: Gx precompute GEMMs (M=HID, N=SEQ, K=DIN) ----
    dim3 gb(TPB);
    dim3 gg(HID / 128, SEQ / 16);
    gru_gemm_bf16<<<gg, gb, 0, stream>>>(Wux, Xb, b_u, Gu, HID, SEQ, DIN);
    gru_gemm_bf16<<<gg, gb, 0, stream>>>(Wrx, Xb, b_r, Gr, HID, SEQ, DIN);
    gru_gemm_bf16<<<gg, gb, 0, stream>>>(Wcx, Xb, b_c, Gc, HID, SEQ, DIN);

    // ---- phase 3: serial recurrence (persistent kernel) ----
    gru_recurrent<<<REC_BLOCKS, TPB_REC, 0, stream>>>(
        Wuh, Wrh, Wch, Gu, Gr, Gc, hbuf, ubuf, rhbuf, Hout, Hbf, counter);

    // ---- phase 4: output projection GEMM (M=DOUT, N=SEQ, K=HID) ----
    dim3 yg(DOUT / 128, SEQ / 16);
    gru_gemm_bf16<<<yg, gb, 0, stream>>>(Wyb, Hbf, b_y, Y, DOUT, SEQ, HID);
}
